// Network_34110630265568
// MI455X (gfx1250) — matrix operand order repaired and checked by reading, not compiled, not run on hardware
//
#include <hip/hip_runtime.h>
#include <hip/hip_bf16.h>
#include <math.h>
#include <stdint.h>

typedef _Float16 f16;
typedef __attribute__((ext_vector_type(16))) _Float16 v16h;
typedef __attribute__((ext_vector_type(8)))  _Float16 v8h;
typedef __attribute__((ext_vector_type(8)))  float    v8f;

#define EPSV 1e-5f

// shift directions (out[h,w] = x[h-dy, w-dx])
__constant__ int c_dy[8] = {1, -1, 0, 0, 1, 1, -1, -1};
__constant__ int c_dx[8] = {0, 0, 1, -1, 1, -1, 1, -1};

// ---------------------------------------------------------------------------
// conv0: 3x3, 3->16, pad 1, stride 1. NCHW input -> channel-major [16][N*H*W]
// ---------------------------------------------------------------------------
__global__ void conv0_k(const float* __restrict__ x, const float* __restrict__ w,
                        float* __restrict__ out, int N, int H, int W) {
  int HW = H * W;
  int P = N * HW;
  int total = 16 * P;
  int idx = blockIdx.x * 256 + threadIdx.x;
  if (idx >= total) return;
  int co = idx / P;
  int p  = idx - co * P;
  int n  = p / HW;
  int hw = p - n * HW;
  int h = hw / W, ww = hw - h * W;
  float s = 0.f;
  for (int ci = 0; ci < 3; ++ci) {
    const float* xb = x + ((size_t)n * 3 + ci) * HW;
    const float* wb = w + (co * 3 + ci) * 9;
    for (int ky = 0; ky < 3; ++ky) {
      int hy = h + ky - 1;
      if (hy < 0 || hy >= H) continue;
      for (int kx = 0; kx < 3; ++kx) {
        int wx = ww + kx - 1;
        if (wx < 0 || wx >= W) continue;
        s += wb[ky * 3 + kx] * xb[hy * W + wx];
      }
    }
  }
  out[idx] = s;
}

// ---------------------------------------------------------------------------
// Per-channel training-mode BN stats: mean + rsqrt(var+eps). One block/channel.
// ---------------------------------------------------------------------------
__global__ void bn_stats_k(const float* __restrict__ X, float* __restrict__ mean,
                           float* __restrict__ rstd, int P) {
  __shared__ float ss[256], sq[256];
  int c = blockIdx.x;
  const float* base = X + (size_t)c * P;
  float s = 0.f, q = 0.f;
  for (int i = threadIdx.x; i < P; i += 256) {
    float v = base[i];
    s += v; q += v * v;
  }
  ss[threadIdx.x] = s; sq[threadIdx.x] = q;
  __syncthreads();
  for (int st = 128; st > 0; st >>= 1) {
    if ((int)threadIdx.x < st) {
      ss[threadIdx.x] += ss[threadIdx.x + st];
      sq[threadIdx.x] += sq[threadIdx.x + st];
    }
    __syncthreads();
  }
  if (threadIdx.x == 0) {
    float m = ss[0] / (float)P;
    float var = sq[0] / (float)P - m * m;
    mean[c] = m;
    rstd[c] = rsqrtf(var + EPSV);
  }
}

// ---------------------------------------------------------------------------
// h = relu(bn(x)) written straight to f16 (GEMM operand format)
// ---------------------------------------------------------------------------
__global__ void bn_act_f16_k(const float* __restrict__ X, const float* __restrict__ g,
                             const float* __restrict__ b, const float* __restrict__ mean,
                             const float* __restrict__ rstd, f16* __restrict__ H,
                             int C, int P) {
  int idx = blockIdx.x * 256 + threadIdx.x;
  if (idx >= C * P) return;
  int c = idx / P;
  float v = (X[idx] - mean[c]) * rstd[c] * g[c] + b[c];
  H[idx] = (f16)(v > 0.f ? v : 0.f);
}

// ---------------------------------------------------------------------------
// stride-2 spatial subsample of an f16 activation tensor [C][N][Hs][Ws]
// ---------------------------------------------------------------------------
__global__ void down2_f16_k(const f16* __restrict__ H, f16* __restrict__ O,
                            int C, int N, int Hs, int Ws) {
  int Ho = Hs >> 1, Wo = Ws >> 1;
  int HWo = Ho * Wo;
  int total = C * N * HWo;
  int idx = blockIdx.x * 256 + threadIdx.x;
  if (idx >= total) return;
  int cn = idx / HWo;
  int hw = idx - cn * HWo;
  int ho = hw / Wo, wo = hw - ho * Wo;
  O[idx] = H[(size_t)cn * Hs * Ws + (size_t)(ho * 2) * Ws + wo * 2];
}

// ---------------------------------------------------------------------------
// Shift-aware BN stats: stats of shift_op(T) computed directly from T.
// Channel group g = c / (C/9): g<8 -> direction DIRS[g], else all-zero channel.
// ---------------------------------------------------------------------------
__global__ void shift_stats_k(const float* __restrict__ T, float* __restrict__ mean,
                              float* __restrict__ rstd, int C, int N, int H, int W) {
  __shared__ float ss[256], sq[256];
  int c = blockIdx.x;
  int cpg = C / 9;
  int g = c / cpg;
  bool live = (g < 8);
  int dy = live ? c_dy[g] : 0;
  int dx = live ? c_dx[g] : 0;
  int HW = H * W;
  int P = N * HW;
  const float* base = T + (size_t)c * P;
  float s = 0.f, q = 0.f;
  for (int i = threadIdx.x; i < P; i += 256) {
    int n = i / HW, hw = i - n * HW;
    int h = hw / W, w = hw - h * W;
    int hs = h - dy, wsx = w - dx;
    float v = 0.f;
    if (live && hs >= 0 && hs < H && wsx >= 0 && wsx < W)
      v = base[n * HW + hs * W + wsx];
    s += v; q += v * v;
  }
  ss[threadIdx.x] = s; sq[threadIdx.x] = q;
  __syncthreads();
  for (int st = 128; st > 0; st >>= 1) {
    if ((int)threadIdx.x < st) {
      ss[threadIdx.x] += ss[threadIdx.x + st];
      sq[threadIdx.x] += sq[threadIdx.x + st];
    }
    __syncthreads();
  }
  if (threadIdx.x == 0) {
    float m = ss[0] / (float)P;
    float var = sq[0] / (float)P - m * m;
    mean[c] = m;
    rstd[c] = rsqrtf(var + EPSV);
  }
}

// ---------------------------------------------------------------------------
// fused shift + BN2 + ReLU + f16 cast (no intermediate shifted tensor)
// ---------------------------------------------------------------------------
__global__ void shift_bn_act_k(const float* __restrict__ T, const float* __restrict__ g,
                               const float* __restrict__ b, const float* __restrict__ mean,
                               const float* __restrict__ rstd, f16* __restrict__ H2,
                               int C, int N, int H, int W) {
  int HW = H * W;
  int P = N * HW;
  int idx = blockIdx.x * 256 + threadIdx.x;
  if (idx >= C * P) return;
  int c = idx / P;
  int i = idx - c * P;
  int cpg = C / 9;
  int gg = c / cpg;
  bool live = (gg < 8);
  int dy = live ? c_dy[gg] : 0;
  int dx = live ? c_dx[gg] : 0;
  int n = i / HW, hw = i - n * HW;
  int h = hw / W, w = hw - h * W;
  int hs = h - dy, wsx = w - dx;
  float v = 0.f;
  if (live && hs >= 0 && hs < H && wsx >= 0 && wsx < W)
    v = T[(size_t)c * P + n * HW + hs * W + wsx];
  float o = (v - mean[c]) * rstd[c] * g[c] + b[c];
  H2[idx] = (f16)(o > 0.f ? o : 0.f);
}

// ---------------------------------------------------------------------------
// simple helpers
// ---------------------------------------------------------------------------
__global__ void copy_f32_k(const float* __restrict__ s, float* __restrict__ d, size_t n) {
  size_t i = (size_t)blockIdx.x * 256 + threadIdx.x;
  if (i < n) d[i] = s[i];
}

__global__ void cvt_f16_k(const float* __restrict__ s, f16* __restrict__ d, int n) {
  int i = blockIdx.x * 256 + threadIdx.x;
  if (i < n) d[i] = (f16)s[i];
}

// ---------------------------------------------------------------------------
// WMMA GEMM: Y[M][P] (+)= W[M][K] * X[K][P], f16 inputs, fp32 accumulate.
// Block: 256 threads = 8 waves (wave32), tile 64(M) x 256(P), K-step 32.
// Waves arranged 2(M) x 4(P); each wave owns 32x64 = 2x4 WMMA fragments
// -> 8 v_wmma_f32_16x16x32_f16 per K-step with A-fragment reuse.
// A tile staged via GLOBAL_LOAD_ASYNC_TO_LDS_B128 (ASYNCcnt / s_wait_asynccnt);
// B tile staged transposed so all fragment reads are ds_load_b128.
// ---------------------------------------------------------------------------
#define BM 64
#define BP 256
#define BK 32
#define BSTR 40  // padded f16 stride for transposed B tile (80B = 16B multiple)

__global__ __launch_bounds__(256, 2)
void gemm_wmma_k(const f16* __restrict__ Wm, const f16* __restrict__ Xm,
                 float* __restrict__ Ym, int M, int K, int P, int accum) {
  __shared__ f16 As[BM * BK];       // row-major [64][32]
  __shared__ f16 Bt[BP * BSTR];     // transposed [256][40]

  const int tid  = threadIdx.x;
  const int m0   = blockIdx.y * BM;
  const int p0   = blockIdx.x * BP;
  const int wave = tid >> 5;
  const int lane = tid & 31;
  const int r    = lane & 15;       // row (A) / column (B,D) within 16
  const int half = lane >> 4;       // K-half select / D row-half select
  const int wm   = wave >> 2;       // 0..1 -> 32-row M strip
  const int wp   = wave & 3;        // 0..3 -> 64-col P strip
  (void)M;

  v8f acc[2][4] = {};

  const int la = tid * 8;
  const int am = la >> 5;           // A tile row   (0..63)
  const int ak = la & 31;           // A tile k off (multiple of 8)
  // LDS byte offset of this thread's A-tile slot (generic ptr low 32 bits ==
  // LDS offset per CDNA5 aperture rule)
  const unsigned ldsA = (unsigned)(size_t)(const void*)(As + am * BK + ak);
  const f16* gA = Wm + (size_t)(m0 + am) * K + ak;

  for (int k0 = 0; k0 < K; k0 += BK) {
    __syncthreads();
    // ---- stage A (64x32 f16): async DMA straight into LDS, 16B per lane ----
    if (k0 + ak + 8 <= K) {
      uint64_t ga = (uint64_t)(const void*)(gA + k0);
      asm volatile("global_load_async_to_lds_b128 %0, %1, off"
                   :: "v"(ldsA), "v"(ga) : "memory");
    } else {
      v8h z = {};
      *(v8h*)(As + am * BK + ak) = z;   // zero-pad K tail (K=16 first block)
    }
    // ---- stage B transposed: Bt[p][k], coalesced b128 global reads ----
#pragma unroll
    for (int u0 = 0; u0 < 4; ++u0) {
      int u  = tid + u0 * 256;
      int bk = u >> 5;              // 0..31
      int bp = (u & 31) * 8;        // 0..248
      v8h vb = {};
      if (k0 + bk < K)
        vb = *(const v8h*)(Xm + (size_t)(k0 + bk) * P + p0 + bp);
#pragma unroll
      for (int j = 0; j < 8; ++j)
        Bt[(bp + j) * BSTR + bk] = vb[j];
    }
    // ---- L2 prefetch of next K tile (global_prefetch_b8) ----
    if (k0 + BK + 31 < K) {
      __builtin_prefetch(gA + k0 + BK, 0, 1);
      __builtin_prefetch(Xm + (size_t)(k0 + BK + lane) * P + p0 + (wave << 5), 0, 1);
    }
    // wave's async A-copy must land in LDS before signaling the barrier
    asm volatile("s_wait_asynccnt 0x0" ::: "memory");
    __syncthreads();

    // ---- fragment loads per CDNA5 wave32 WMMA VGPR layout ----
    union AB { v16h v; v8h h[2]; };
    AB a[2], b[4];
#pragma unroll
    for (int i = 0; i < 2; ++i) {
      // A 16x32: lanes 0-15 hold K={0..7,16..23}, lanes 16-31 K={8..15,24..31}
      const f16* ar = As + (wm * 32 + i * 16 + r) * BK;
      a[i].h[0] = *(const v8h*)(ar + half * 8);
      a[i].h[1] = *(const v8h*)(ar + 16 + half * 8);
    }
#pragma unroll
    for (int j = 0; j < 4; ++j) {
      // B 32x16: lane = column, lanes 0-15 K={0..7,16..23}, lanes 16-31 K={8..15,24..31}
      const f16* br = Bt + (wp * 64 + j * 16 + r) * BSTR + half * 8;
      b[j].h[0] = *(const v8h*)(br);
      b[j].h[1] = *(const v8h*)(br + 16);
    }
#pragma unroll
    for (int i = 0; i < 2; ++i)
#pragma unroll
      for (int j = 0; j < 4; ++j)
        acc[i][j] = __builtin_amdgcn_wmma_f32_16x16x32_f16(
            false, a[i].v, false, b[j].v, (short)0, acc[i][j], false, false);
  }

  // ---- epilogue: D VGPR v -> row half*8+v, col = r (per ISA C/D layout) ----
  const int rowb = m0 + wm * 32 + half * 8;
  const int colb = p0 + wp * 64 + r;
#pragma unroll
  for (int i = 0; i < 2; ++i) {
#pragma unroll
    for (int j = 0; j < 4; ++j) {
      int row = rowb + i * 16;
      int col = colb + j * 16;
      if (accum) {
#pragma unroll
        for (int v = 0; v < 8; ++v)
          Ym[(size_t)(row + v) * P + col] += acc[i][j][v];
      } else {
#pragma unroll
        for (int v = 0; v < 8; ++v)
          Ym[(size_t)(row + v) * P + col] = acc[i][j][v];
      }
    }
  }
}

// ---------------------------------------------------------------------------
// final BN + ReLU + global average pool: pooled[n][c]
// ---------------------------------------------------------------------------
__global__ void bnpool_k(const float* __restrict__ X, const float* __restrict__ g,
                         const float* __restrict__ b, const float* __restrict__ mean,
                         const float* __restrict__ rstd, float* __restrict__ pooled,
                         int C, int N, int HW) {
  int idx = blockIdx.x * 256 + threadIdx.x;
  if (idx >= N * C) return;
  int n = idx / C;
  int c = idx - n * C;
  const float* base = X + ((size_t)c * N + n) * HW;
  float m = mean[c], rs = rstd[c], gg = g[c], bb = b[c];
  float s = 0.f;
  for (int i = 0; i < HW; ++i) {
    float v = (base[i] - m) * rs * gg + bb;
    s += (v > 0.f ? v : 0.f);
  }
  pooled[idx] = s / (float)HW;
}

// ---------------------------------------------------------------------------
// FC 512->10 + log_softmax, one wave per sample
// ---------------------------------------------------------------------------
__global__ void head_k(const float* __restrict__ pooled, const float* __restrict__ fw,
                       const float* __restrict__ fb, float* __restrict__ out, int C) {
  __shared__ float lg[10];
  __shared__ float mv, ls;
  int n = blockIdx.x, t = threadIdx.x;
  if (t < 10) {
    float s = fb[t];
    const float* pv = pooled + (size_t)n * C;
    const float* wv = fw + (size_t)t * C;
    for (int c = 0; c < C; ++c) s += pv[c] * wv[c];
    lg[t] = s;
  }
  __syncthreads();
  if (t == 0) {
    float m = lg[0];
    for (int j = 1; j < 10; ++j) m = fmaxf(m, lg[j]);
    float s = 0.f;
    for (int j = 0; j < 10; ++j) s += __expf(lg[j] - m);
    mv = m; ls = __logf(s);
  }
  __syncthreads();
  if (t < 10) out[n * 10 + t] = lg[t] - mv - ls;
}

// ---------------------------------------------------------------------------
// host orchestration
// ---------------------------------------------------------------------------
extern "C" void kernel_launch(void* const* d_in, const int* in_sizes, int n_in,
                              void* d_out, int out_size, void* d_ws, size_t ws_size,
                              hipStream_t stream) {
  (void)in_sizes; (void)n_in; (void)out_size; (void)ws_size;

  // Inputs in setup_inputs() insertion order:
  // x, conv0_w, then per block: bn1_g, bn1_b, conv1_w, bn2_g, bn2_b, conv3_w,
  // [convs_w if ip!=op], then bnf_g, bnf_b, fc_w, fc_b
  int ii = 0;
  const float* x  = (const float*)d_in[ii++];
  const float* w0 = (const float*)d_in[ii++];
  const int IPs[6] = {16, 128, 128, 256, 256, 512};
  const int OPs[6] = {128, 128, 256, 256, 512, 512};
  const int STs[6] = {1, 1, 2, 1, 2, 1};
  const float *bn1g[6], *bn1b[6], *w1[6], *bn2g[6], *bn2b[6], *w3[6], *wsw[6];
  for (int bI = 0; bI < 6; ++bI) {
    bn1g[bI] = (const float*)d_in[ii++];
    bn1b[bI] = (const float*)d_in[ii++];
    w1[bI]   = (const float*)d_in[ii++];
    bn2g[bI] = (const float*)d_in[ii++];
    bn2b[bI] = (const float*)d_in[ii++];
    w3[bI]   = (const float*)d_in[ii++];
    wsw[bI]  = (IPs[bI] != OPs[bI]) ? (const float*)d_in[ii++] : nullptr;
  }
  const float* bnfg = (const float*)d_in[ii++];
  const float* bnfb = (const float*)d_in[ii++];
  const float* fcw  = (const float*)d_in[ii++];
  const float* fcb  = (const float*)d_in[ii++];

  // workspace carve-out (max live tensor: 128ch x 131072 positions)
  char* wsb = (char*)d_ws;
  size_t off = 0;
  auto alloc = [&](size_t bytes) -> void* {
    void* p = wsb + off;
    off = (off + bytes + 255) & ~(size_t)255;
    return p;
  };
  const size_t PMAX = 131072;
  const size_t EMAX = 128 * PMAX;             // 16.78M elems covers all layers
  float* Xb  = (float*)alloc(EMAX * sizeof(float));
  float* Yb  = (float*)alloc(EMAX * sizeof(float));
  float* T0  = (float*)alloc(EMAX * sizeof(float));
  f16*   Hf  = (f16*)alloc(EMAX * sizeof(f16));
  f16*   H2f = (f16*)alloc(EMAX * sizeof(f16));
  f16*   HSf = (f16*)alloc((EMAX / 4) * sizeof(f16));
  f16*   Wf  = (f16*)alloc(512 * 512 * sizeof(f16));
  float* mean1 = (float*)alloc(512 * sizeof(float));
  float* rstd1 = (float*)alloc(512 * sizeof(float));
  float* mean2 = (float*)alloc(512 * sizeof(float));
  float* rstd2 = (float*)alloc(512 * sizeof(float));
  float* pooled = (float*)alloc(128 * 512 * sizeof(float));

  const int Nimg = 128;
  int Hc = 32, Wc = 32;
  int Cin = 16;
  int P = Nimg * Hc * Wc;

  // stem
  {
    int total = 16 * P;
    conv0_k<<<(total + 255) / 256, 256, 0, stream>>>(x, w0, Xb, Nimg, Hc, Wc);
  }

  float* Xcur = Xb;
  float* Ycur = Yb;
  for (int bI = 0; bI < 6; ++bI) {
    int ip = IPs[bI], opc = OPs[bI], st = STs[bI];

    // BN1 + ReLU -> f16
    bn_stats_k<<<Cin, 256, 0, stream>>>(Xcur, mean1, rstd1, P);
    {
      int tot = Cin * P;
      bn_act_f16_k<<<(tot + 255) / 256, 256, 0, stream>>>(
          Xcur, bn1g[bI], bn1b[bI], mean1, rstd1, Hf, Cin, P);
    }

    // stride handling: one downsample feeds both conv1 and shortcut conv
    const f16* Gin = Hf;
    int Ho = Hc, Wo = Wc, Po = P;
    if (st == 2) {
      Ho = Hc / 2; Wo = Wc / 2; Po = P / 4;
      int tot = Cin * Po;
      down2_f16_k<<<(tot + 255) / 256, 256, 0, stream>>>(Hf, HSf, Cin, Nimg, Hc, Wc);
      Gin = HSf;
    }

    // shortcut -> Ycur
    if (ip == opc) {
      size_t tot = (size_t)opc * Po;
      copy_f32_k<<<(unsigned)((tot + 255) / 256), 256, 0, stream>>>(Xcur, Ycur, tot);
    } else {
      int nw = opc * ip;
      cvt_f16_k<<<(nw + 255) / 256, 256, 0, stream>>>(wsw[bI], Wf, nw);
      dim3 g(Po / BP, opc / BM);
      gemm_wmma_k<<<g, 256, 0, stream>>>(Wf, Gin, Ycur, opc, ip, Po, 0);
    }

    // conv1 (1x1) -> T0
    {
      int nw = opc * ip;
      cvt_f16_k<<<(nw + 255) / 256, 256, 0, stream>>>(w1[bI], Wf, nw);
      dim3 g(Po / BP, opc / BM);
      gemm_wmma_k<<<g, 256, 0, stream>>>(Wf, Gin, T0, opc, ip, Po, 0);
    }

    // fused shift + BN2 + ReLU -> f16
    shift_stats_k<<<opc, 256, 0, stream>>>(T0, mean2, rstd2, opc, Nimg, Ho, Wo);
    {
      int tot = opc * Po;
      shift_bn_act_k<<<(tot + 255) / 256, 256, 0, stream>>>(
          T0, bn2g[bI], bn2b[bI], mean2, rstd2, H2f, opc, Nimg, Ho, Wo);
    }

    // conv3 (1x1) accumulates into shortcut: Ycur += W3 * H2
    {
      int nw = opc * opc;
      cvt_f16_k<<<(nw + 255) / 256, 256, 0, stream>>>(w3[bI], Wf, nw);
      dim3 g(Po / BP, opc / BM);
      gemm_wmma_k<<<g, 256, 0, stream>>>(Wf, H2f, Ycur, opc, opc, Po, 1);
    }

    // ping-pong
    float* t = Xcur; Xcur = Ycur; Ycur = t;
    Cin = opc; P = Po; Hc = Ho; Wc = Wo;
  }

  // final BN + ReLU + pool + FC + log_softmax
  bn_stats_k<<<Cin, 256, 0, stream>>>(Xcur, mean1, rstd1, P);
  {
    int tot = Nimg * Cin;
    bnpool_k<<<(tot + 255) / 256, 256, 0, stream>>>(
        Xcur, bnfg, bnfb, mean1, rstd1, pooled, Cin, Nimg, Hc * Wc);
  }
  head_k<<<Nimg, 32, 0, stream>>>(pooled, fcw, fcb, (float*)d_out, Cin);
}
